// GraphConvNet_85186381349438
// MI455X (gfx1250) — matrix operand order repaired and checked by reading, not compiled, not run on hardware
//
#include <hip/hip_runtime.h>
#include <hip/hip_bf16.h>

typedef _Float16 f16;
typedef f16   v8h  __attribute__((ext_vector_type(8)));
typedef f16   v16h __attribute__((ext_vector_type(16)));
typedef float v8f  __attribute__((ext_vector_type(8)));

#define COLS 224          // 200 padded to 7*32 (col 200 = homogeneous 1.0 lane)
#define ROWS 32           // nodes per wave (two 16-node D tiles sharing A fragments)
#define WAVES 8
#define NODES_PER_BLOCK (WAVES * ROWS)   // 256
#define NLAYERS 6
#define KCH 7             // 224/32 K-chunks
#define NTI 14            // 224/16 N-tiles (output-column tiles)
#define NFRAG (NTI * KCH)                // 98 fragments per layer
#define PF 4              // A-fragment prefetch depth (4 pairs of b128 in flight)
#define FRAG_WORDS_PER_LAYER (NFRAG * 32 * 8)   // 25088 dwords = 100352 B

// ---------------- workspace layout (byte offsets into d_ws) ----------------
#define OFF_H      0              // float[200000] : agg, then h (in place)
#define OFF_FRAG   800768         // uint32[6*25088] pre-swizzled f16 A-fragments (W^T), 32B aligned
#define OFF_SMALL  1402880        // float[672]: w_in(224) b_in(224, [200]=1) w_out(224, [200]=b_out)

// ---------------------------------------------------------------------------
// 1) edge scatter: agg[dst] += x[src]
__global__ void k_scatter(const long long* __restrict__ ei,
                          const float* __restrict__ x,
                          float* __restrict__ agg, int nE) {
    int e = blockIdx.x * blockDim.x + threadIdx.x;
    if (e < nE) {
        int s = (int)ei[e];
        int d = (int)ei[nE + e];
        atomicAdd(&agg[d], x[s]);
    }
}

// 2) h = agg*w_rel + b_rel + x*w_root (overwrite agg in place)
__global__ void k_node_h(float* __restrict__ agg, const float* __restrict__ x,
                         const float* __restrict__ w_rel, const float* __restrict__ b_rel,
                         const float* __restrict__ w_root, int n) {
    int i = blockIdx.x * blockDim.x + threadIdx.x;
    if (i < n) {
        agg[i] = agg[i] * w_rel[0] + b_rel[0] + x[i] * w_root[0];
    }
}

// 3) pre-swizzle hidden-layer weights (TRANSPOSED, bias folded in as row K=200)
//    into WMMA A-fragment order (f16, zero-pad).
//      Wpad[k][n] = W[k][n]     (k<200, n<200)
//                 = b_hid[l][n] (k==200, n<200)   <- bias row
//                 = 1           (k==200, n==200)  <- keeps ones-lane alive
//                 = 0           otherwise
//    ISA 16-bit A layout: lane{half, mA}; dword p: kk = (p>>2)*16 + half*8 + (p&3)*2 (+1)
//    frag index order: [l][nt][kc][lane][p]
__global__ void k_prep_frag(const float* __restrict__ wh, const float* __restrict__ bh,
                            unsigned* __restrict__ frag) {
    int t = blockIdx.x * blockDim.x + threadIdx.x;
    if (t >= NLAYERS * FRAG_WORDS_PER_LAYER) return;
    int l = t / FRAG_WORDS_PER_LAYER; int r = t % FRAG_WORDS_PER_LAYER;
    int nt = r / (KCH * 256); r %= (KCH * 256);
    int kc = r / 256;         r %= 256;
    int lane = r / 8;         int p = r % 8;
    int mA   = lane & 15;
    int half = lane >> 4;
    int n  = nt * 16 + mA;
    int kk = ((p >> 2) << 4) + (half << 3) + ((p & 3) << 1);
    int k0 = kc * 32 + kk;
    float lo, hi;
    #pragma unroll
    for (int q = 0; q < 2; ++q) {
        int k = k0 + q;
        float v = 0.f;
        if (n < 200) {
            if (k < 200)       v = wh[l * 40000 + k * 200 + n];
            else if (k == 200) v = bh[l * 200 + n];
        } else if (n == 200 && k == 200) {
            v = 1.f;
        }
        if (q == 0) lo = v; else hi = v;
    }
    union { f16 h[2]; unsigned u; } pk;
    pk.h[0] = (f16)lo; pk.h[1] = (f16)hi;
    frag[t] = pk.u;
}

// 4) zero-padded float copies of small params (with homogeneous-coordinate tricks)
__global__ void k_prep_small(const float* __restrict__ w_in, const float* __restrict__ b_in,
                             const float* __restrict__ w_out, const float* __restrict__ b_out,
                             float* __restrict__ sm) {
    int i = blockIdx.x * blockDim.x + threadIdx.x;
    if (i >= 672) return;
    float v;
    if (i < 224) {                      // w_in padded; [200]=0 so relu(0*h+1)=1 seeds ones-lane
        v = (i < 200) ? w_in[i] : 0.f;
    } else if (i < 448) {               // b_in padded; [200]=1.0
        int c = i - 224;
        v = (c < 200) ? b_in[c] : (c == 200 ? 1.f : 0.f);
    } else {                            // w_out padded; [200]=b_out folds final bias
        int c = i - 448;
        v = (c < 200) ? w_out[c] : (c == 200 ? b_out[0] : 0.f);
    }
    sm[i] = v;
}

// ---------------------------------------------------------------------------
// 5) fused 8-layer MLP. One wave owns 32 nodes (two 16-node tiles) in a
//    private 32x224 f16 LDS slab. Hidden layers: D = A(W^T tile) x B(act^T),
//    bias via homogeneous row; each streamed weight fragment feeds 2 WMMAs.
//    A-fragment stream is explicitly software-pipelined 4 deep.
__global__ void __launch_bounds__(256)
k_mlp(const float* __restrict__ h, const unsigned* __restrict__ wfrag,
      const float* __restrict__ sm, float* __restrict__ out, int nNodes) {
    __shared__ __align__(32) f16 act[WAVES][ROWS * COLS];   // 112 KB

    const int wave    = threadIdx.x >> 5;
    const int lane    = threadIdx.x & 31;
    const int m       = lane & 15;      // node selector within tile (B/D cols)
    const int halfsel = lane >> 4;
    const int node0   = blockIdx.x * NODES_PER_BLOCK + wave * ROWS;

    const float* winf  = sm;
    const float* binf  = sm + 224;
    const float* woutf = sm + 448;

    // ---- layer 0: expansion relu(h_i * w_in + b_in); col 200 becomes 1.0 ----
    for (int c = lane; c < COLS; c += 32) {
        float wv = winf[c], bv = binf[c];
        #pragma unroll
        for (int r = 0; r < ROWS; ++r) {
            int node = node0 + r;
            float hv = (node < nNodes) ? h[node] : 0.f;
            act[wave][r * COLS + c] = (f16)fmaxf(hv * wv + bv, 0.f);
        }
    }
    __syncthreads();

    // ---- 6 hidden layers ----
    for (int l = 0; l < NLAYERS; ++l) {
        // B fragments (act^T, 32x16) for both node tiles: lane{half, node=m};
        // dword v holds K = (v>>2)*16 + half*8 + 2(v&3) (+1) -> two 16-byte runs per lane.
        v16h bfr[2][KCH];
        #pragma unroll
        for (int t = 0; t < 2; ++t) {
            const f16* bbase = &act[wave][(t * 16 + m) * COLS] + (halfsel << 3);
            #pragma unroll
            for (int kc = 0; kc < KCH; ++kc) {
                v8h lo = *(const v8h*)(bbase + kc * 32);
                v8h hi = *(const v8h*)(bbase + kc * 32 + 16);
                bfr[t][kc] = __builtin_shufflevector(lo, hi, 0, 1, 2, 3, 4, 5, 6, 7,
                                                     8, 9, 10, 11, 12, 13, 14, 15);
            }
        }
        const unsigned* wl = wfrag + l * FRAG_WORDS_PER_LAYER + lane * 8;
        if (l + 1 < NLAYERS)
            __builtin_prefetch(wfrag + (l + 1) * FRAG_WORDS_PER_LAYER + lane * 8, 0, 1);

        // prime the 4-deep A-fragment pipeline (fragments are linear in (nt,kc))
        v16h apf[PF];
        #pragma unroll
        for (int i = 0; i < PF; ++i)
            apf[i] = *((const v16h*)(wl + i * 256));

        #pragma unroll
        for (int nt = 0; nt < NTI; ++nt) {
            v8f acc0 = {};  // zero C -> inline SRC2=0; bias via weight row K=200
            v8f acc1 = {};
            #pragma unroll
            for (int kc = 0; kc < KCH; ++kc) {
                const int fi  = nt * KCH + kc;       // linear fragment index
                const int slot = fi % PF;            // constant under full unroll
                v16h aw = apf[slot];
                if (fi + PF < NFRAG)                 // keep 4 pairs in flight
                    apf[slot] = *((const v16h*)(wl + (fi + PF) * 256));
                acc0 = __builtin_amdgcn_wmma_f32_16x16x32_f16(
                        false, aw, false, bfr[0][kc], (short)0, acc0, false, false);
                acc1 = __builtin_amdgcn_wmma_f32_16x16x32_f16(
                        false, aw, false, bfr[1][kc], (short)0, acc1, false, false);
            }
            // D: lane{half, node=m}; VGPR r -> outcol nt*16 + 8*half + r.
            // 8 consecutive f16 -> pack + single ds_store_b128 per tile.
            v8h dv0, dv1;
            #pragma unroll
            for (int r = 0; r < 8; ++r) {
                dv0[r] = (f16)fmaxf(acc0[r], 0.f);
                dv1[r] = (f16)fmaxf(acc1[r], 0.f);
            }
            int cofs = nt * 16 + (halfsel << 3);
            *(v8h*)(&act[wave][m * COLS + cofs])          = dv0;
            *(v8h*)(&act[wave][(16 + m) * COLS + cofs])   = dv1;
        }
        __syncthreads();
    }

    // ---- output layer: sigmoid(act . w_out)  (bias folded at col 200) ----
    int c0 = halfsel * 112;
    #pragma unroll
    for (int t = 0; t < 2; ++t) {
        float s = 0.f;
        const f16* arow = &act[wave][(t * 16 + m) * COLS];
        #pragma unroll 4
        for (int c = 0; c < 112; ++c)
            s += (float)arow[c0 + c] * woutf[c0 + c];
        s += __shfl_xor(s, 16, 32);
        if (halfsel == 0) {
            int node = node0 + t * 16 + m;
            if (node < nNodes) {
                out[node] = 1.f / (1.f + __expf(-s));
            }
        }
    }
}

// ---------------------------------------------------------------------------
extern "C" void kernel_launch(void* const* d_in, const int* in_sizes, int n_in,
                              void* d_out, int out_size, void* d_ws, size_t ws_size,
                              hipStream_t stream) {
    const float*     x      = (const float*)d_in[0];
    const long long* ei     = (const long long*)d_in[1];
    const float*     w_rel  = (const float*)d_in[2];
    const float*     b_rel  = (const float*)d_in[3];
    const float*     w_root = (const float*)d_in[4];
    const float*     w_in   = (const float*)d_in[5];
    const float*     b_in   = (const float*)d_in[6];
    const float*     w_hid  = (const float*)d_in[7];
    const float*     b_hid  = (const float*)d_in[8];
    const float*     w_out  = (const float*)d_in[9];
    const float*     b_out  = (const float*)d_in[10];
    float*           out    = (float*)d_out;

    const int nNodes = in_sizes[0];
    const int nEdges = in_sizes[1] / 2;

    char* ws = (char*)d_ws;
    float*    agg   = (float*)(ws + OFF_H);
    unsigned* wfrag = (unsigned*)(ws + OFF_FRAG);
    float*    sm    = (float*)(ws + OFF_SMALL);

    // aggregation
    hipMemsetAsync(agg, 0, (size_t)nNodes * sizeof(float), stream);
    k_scatter<<<(nEdges + 255) / 256, 256, 0, stream>>>(ei, x, agg, nEdges);
    k_node_h<<<(nNodes + 255) / 256, 256, 0, stream>>>(agg, x, w_rel, b_rel, w_root, nNodes);

    // weight prep (cheap; L2-resident thereafter)
    const int nfrag = NLAYERS * FRAG_WORDS_PER_LAYER;
    k_prep_frag<<<(nfrag + 255) / 256, 256, 0, stream>>>(w_hid, b_hid, wfrag);
    k_prep_small<<<3, 256, 0, stream>>>(w_in, b_in, w_out, b_out, sm);

    // fused MLP
    int blocks = (nNodes + NODES_PER_BLOCK - 1) / NODES_PER_BLOCK;
    k_mlp<<<blocks, 256, 0, stream>>>(agg, wfrag, sm, out, nNodes);
    (void)n_in; (void)out_size; (void)ws_size;
}
